// RNNTLossNumba_13073880449271
// MI455X (gfx1250) — compile-verified, matrix operand order checked by reading, not run on hardware
//
#include <hip/hip_runtime.h>
#include <hip/hip_bf16.h>

// RNN-T loss (mean) for B=16, T=128, U=64, V=1024, fp32.
// Phase 1 (dominant, ~512MB streamed): per-row logsumexp over V=1024 using
//   v_wmma_f32_16x16x32_f16 with an all-ones B matrix as the row-sum engine
//   (co-executes with the v_exp_f32 VALU/TRANS work), plus gather of the
//   blank / label logits.
// Phase 2: log-semiring associative scan over U (wave32 shuffle scan),
//   sequential over T, one wave per batch element.
// Phase 3: mean reduction to d_out[0].

#define B_N 16
#define T_N 128
#define U_N 64
#define V_N 1024
#define BTU (B_N * T_N * U_N)
#define NEGINF (-1e30f)

typedef __attribute__((ext_vector_type(16))) _Float16 v16h;
typedef __attribute__((ext_vector_type(8)))  float    v8f;

__device__ __forceinline__ float max4_(float4 v) {
    return fmaxf(fmaxf(v.x, v.y), fmaxf(v.z, v.w));
}

__device__ __forceinline__ float logaddexp_(float a, float b) {
    float m = fmaxf(a, b);
    float n = fminf(a, b);
    return m + log1pf(__expf(n - m));
}

// ---------------------------------------------------------------------------
// Kernel 1: fused log_softmax (logsumexp) + gather.
// One wave per 16 consecutive (b,t,u) rows. Rows map to WMMA M dimension.
// Pass A: row max (stream from HBM). Pass B: exp(x - max), pack to f16,
// row-sum via WMMA against all-ones B (K=32 per WMMA, 32 chunks per row).
// Lane L (and L+16) own row (L&15); lane L covers columns
// chunk*32 + (L>>4)*16 .. +15, so each element is consumed exactly once and
// the K-order inside a chunk is irrelevant (B is all ones).
// ---------------------------------------------------------------------------
__global__ __launch_bounds__(256, 2) void rnnt_logsoftmax_gather(
    const float* __restrict__ acts, const int* __restrict__ labels,
    float* __restrict__ lpb, float* __restrict__ ews)
{
    __shared__ float s_sum[8][16];
    const int    lane = threadIdx.x & 31;
    const int    wv   = threadIdx.x >> 5;
    const size_t row0 = ((size_t)blockIdx.x * 8 + wv) * 16;
    const size_t row  = row0 + (size_t)(lane & 15);
    const float* rp   = acts + row * V_N + (size_t)((lane >> 4) * 16);

    // ---- pass A: row max -------------------------------------------------
    float mx = -3.4e38f;
#pragma unroll 8
    for (int ch = 0; ch < 32; ++ch) {
        const float4* p = (const float4*)(rp + ch * 32);
        float4 q0 = p[0], q1 = p[1], q2 = p[2], q3 = p[3];
        mx = fmaxf(mx, fmaxf(fmaxf(max4_(q0), max4_(q1)),
                             fmaxf(max4_(q2), max4_(q3))));
    }
    mx = fmaxf(mx, __shfl_xor(mx, 16));   // lanes L, L+16 share a row

    // ---- pass B: exp + WMMA row-sum -------------------------------------
    v16h ones;
#pragma unroll
    for (int i = 0; i < 16; ++i) ones[i] = (_Float16)1.0f;

    v8f c = {0.f, 0.f, 0.f, 0.f, 0.f, 0.f, 0.f, 0.f};

#pragma unroll 4
    for (int ch = 0; ch < 32; ++ch) {
        const float4* p = (const float4*)(rp + ch * 32);
        float4 q0 = p[0], q1 = p[1], q2 = p[2], q3 = p[3];
        float x[16];
        x[0]=q0.x; x[1]=q0.y; x[2]=q0.z;  x[3]=q0.w;
        x[4]=q1.x; x[5]=q1.y; x[6]=q1.z;  x[7]=q1.w;
        x[8]=q2.x; x[9]=q2.y; x[10]=q2.z; x[11]=q2.w;
        x[12]=q3.x;x[13]=q3.y;x[14]=q3.z; x[15]=q3.w;
        v16h a;
#pragma unroll
        for (int i = 0; i < 16; ++i)
            a[i] = (_Float16)__expf(x[i] - mx);   // in (0,1], f16-safe
        // D = A(16x32 f16) * ones(32x16) + C : accumulates per-row sum into C
        c = __builtin_amdgcn_wmma_f32_16x16x32_f16(
                false, a, false, ones, (short)0, c, false, false);
    }

    // D columns are identical (B==ones): lane 0 holds rows 0..7 sums in
    // c[0..7], lane 16 holds rows 8..15. Stage through LDS.
    if (lane == 0) {
#pragma unroll
        for (int j = 0; j < 8; ++j) s_sum[wv][j] = c[j];
    } else if (lane == 16) {
#pragma unroll
        for (int j = 0; j < 8; ++j) s_sum[wv][8 + j] = c[j];
    }
    __syncthreads();

    if (lane < 16) {
        float lse = __logf(s_sum[wv][lane]) + mx;
        float x0  = acts[row * V_N];                    // BLANK = 0
        lpb[row]  = x0 - lse;
        int u = (int)(row & (U_N - 1));
        int b = (int)(row >> 13);                       // / (T*U) = / 8192
        if (u < U_N - 1) {
            int lbl = labels[b * (U_N - 1) + u];
            ews[row + 1] = acts[row * V_N + (size_t)lbl] - lse;  // e[u+1]
        }
        if (u == 0) ews[row] = NEGINF;                  // e[0] = -inf
    }
}

// ---------------------------------------------------------------------------
// Kernel 2: alpha recurrence. One wave per batch element, 2 u's per lane.
// Per t: alpha[u] = logaddexp(alpha_prev[u]+lpb[t-1,u], alpha[u-1]+e[t,u])
// solved with the log-semiring associative scan:
//   combine((e1,c1),(e2,c2)) = (e1+e2, logaddexp(c2, c1+e2))
// ---------------------------------------------------------------------------
__global__ __launch_bounds__(32) void rnnt_alpha_scan(
    const float* __restrict__ lpb, const float* __restrict__ ews,
    float* __restrict__ ll)
{
    const int    b    = blockIdx.x;
    const int    L    = threadIdx.x;          // lane, owns u = 2L, 2L+1
    const size_t base = (size_t)b * T_N * U_N;

    // t = 0: alpha0[u] = prefix-sum of label log-probs (e shifted, e[0]->0)
    float v0 = (L == 0) ? 0.f : ews[base + 2 * L];
    float v1 = ews[base + 2 * L + 1];
    float S  = v0 + v1;
#pragma unroll
    for (int d = 1; d < 32; d <<= 1) {
        float tS = __shfl_up(S, d);
        if (L >= d) S += tS;
    }
    float Sp = __shfl_up(S, 1);
    if (L == 0) Sp = 0.f;
    float a0 = Sp + v0;   // alpha[2L]
    float a1 = S;         // alpha[2L+1]

    for (int t = 1; t < T_N; ++t) {
        float2 pb = *(const float2*)(lpb + base + (size_t)(t - 1) * U_N + 2 * L);
        float2 ee = *(const float2*)(ews + base + (size_t)t * U_N + 2 * L);
        float cA = a0 + pb.x;
        float cB = a1 + pb.y;
        float e0 = ee.x, e1 = ee.y;

        // combine the lane's two elements
        float le = e0 + e1;
        float lc = logaddexp_(cB, cA + e1);

        // inclusive shuffle scan across 32 lanes
#pragma unroll
        for (int d = 1; d < 32; d <<= 1) {
            float pe = __shfl_up(le, d);
            float pc = __shfl_up(lc, d);
            if (L >= d) {
                lc = logaddexp_(lc, pc + le);   // uses old le
                le = pe + le;
            }
        }
        float Spc = __shfl_up(lc, 1);
        if (L == 0) Spc = NEGINF;               // scan identity (0, -inf)
        a0 = logaddexp_(cA, Spc + e0);          // alpha[2L]
        a1 = lc;                                // alpha[2L+1]
    }
    if (L == 31)
        ll[b] = a1 + lpb[base + (size_t)(T_N - 1) * U_N + (U_N - 1)];
}

// ---------------------------------------------------------------------------
// Kernel 3: out[0] = mean of costs = sum(-loglike) / B
// ---------------------------------------------------------------------------
__global__ __launch_bounds__(32) void rnnt_finalize(
    const float* __restrict__ ll, float* __restrict__ out)
{
    int L = threadIdx.x;
    float v = (L < B_N) ? ll[L] : 0.f;
#pragma unroll
    for (int d = 16; d > 0; d >>= 1) v += __shfl_xor(v, d);
    if (L == 0) out[0] = -v / (float)B_N;
}

extern "C" void kernel_launch(void* const* d_in, const int* in_sizes, int n_in,
                              void* d_out, int out_size, void* d_ws, size_t ws_size,
                              hipStream_t stream) {
    (void)in_sizes; (void)n_in; (void)out_size; (void)ws_size;
    const float* acts   = (const float*)d_in[0];
    const int*   labels = (const int*)d_in[1];
    // d_in[2]/d_in[3] (act_lens/label_lens) are full per the reference.

    float* wsf = (float*)d_ws;
    float* lpb = wsf;              // [B,T,U]   log P(blank)
    float* ews = wsf + BTU;        // [B,T,U]   e[0]=-inf, e[u]=lp_label[u-1]
    float* ll  = wsf + 2 * BTU;    // [B]       log-likelihoods

    const int waves_per_block = 8;                       // 256 threads
    const int rows_per_wave   = 16;
    const int nblocks = BTU / (waves_per_block * rows_per_wave);  // 1024

    rnnt_logsoftmax_gather<<<nblocks, 256, 0, stream>>>(acts, labels, lpb, ews);
    rnnt_alpha_scan<<<B_N, 32, 0, stream>>>(lpb, ews, ll);
    rnnt_finalize<<<1, 32, 0, stream>>>(ll, (float*)d_out);
}
